// MixtureOfExperts_74234214744418
// MI455X (gfx1250) — compile-verified, matrix operand order checked by reading, not run on hardware
//
#include <hip/hip_runtime.h>
#include <hip/hip_bf16.h>

// ---------------- problem constants ----------------
#define D_MODEL   1024
#define D_HIDDEN  2048
#define N_EXPERTS 8
#define TOP_K     2
#define N_TOKENS  8192              // B*T = 4*2048
#define N_PAIRS   (N_TOKENS * TOP_K)
#define TILE_M    32
#define HK        64                // hidden-dim chunk processed per iteration
#define KS        512               // K rows staged per GEMM1 phase
#define WPAD      8                 // LDS stride pad (bank decorrelation)
#define MAXPAIRS  (N_PAIRS + N_EXPERTS * TILE_M)   // 16640
#define MAXTILES  (MAXPAIRS / TILE_M)              // 520

// LDS layout (bytes)
#define XS_STRIDE (D_MODEL + WPAD)              // 1032 bf16
#define HS_STRIDE (HK + WPAD)                   // 72 bf16
#define W12_STRIDE (KS + WPAD)                  // 520 bf16
#define W3_STRIDE  (HK + WPAD)                  // 72 bf16
#define LDS_XS   0
#define LDS_WT   (TILE_M * XS_STRIDE * 2)                      // 66048
#define LDS_W2T  (LDS_WT + HK * W12_STRIDE * 2)                // + 66560
#define LDS_HS   (LDS_WT + D_MODEL * W3_STRIDE * 2)            // union: W3T is largest
#define LDS_TOTAL (LDS_HS + TILE_M * HS_STRIDE * 2)            // ~213 KB (<320 KB WGP)

typedef __attribute__((ext_vector_type(16))) __bf16          v16bf;
typedef __attribute__((ext_vector_type(8)))  float           v8f;
typedef __attribute__((ext_vector_type(16))) unsigned short  u16x16;
typedef __attribute__((ext_vector_type(4)))  unsigned short  u16x4;
typedef __attribute__((ext_vector_type(4)))  float           f32x4;

// ---------------- helpers ----------------
__device__ __forceinline__ unsigned short f2bf(float f) {
    unsigned int u = __float_as_uint(f);
    unsigned int r = u + 0x7FFFu + ((u >> 16) & 1u);   // round-to-nearest-even
    return (unsigned short)(r >> 16);
}

__device__ __forceinline__ v8f wmma_bf16(v16bf a, v16bf b, v8f c) {
    // emits v_wmma_f32_16x16x32_bf16
    return __builtin_amdgcn_wmma_f32_16x16x32_bf16(
        false, a, false, b, (short)0, c, false, false);
}

// 16x32 bf16 fragment from an LDS matrix whose "fragment rows" are contiguous
// in K (A matrices row-major; B matrices staged transposed). CDNA5 layout:
//   lane<16 : row=lane,    K in {k0..k0+7, k0+16..k0+23}
//   lane>=16: row=lane-16,  K in {k0+8..k0+15, k0+24..k0+31}
__device__ __forceinline__ v16bf frag_lds(const unsigned short* base, int stride,
                                          int row, int k0, int hi) {
    u16x16 u;
    const unsigned short* p = base + row * stride + k0 + hi * 8;
#pragma unroll
    for (int j = 0; j < 8; ++j) u[j]     = p[j];        // 16B ds load
#pragma unroll
    for (int j = 0; j < 8; ++j) u[8 + j] = p[16 + j];   // 16B ds load
    return __builtin_bit_cast(v16bf, u);
}

// ---------------- kernel 1: init ----------------
__global__ void moe_init_k(float* __restrict__ out, int out_n,
                           int* __restrict__ ptok, float* __restrict__ pprob,
                           int* __restrict__ meta, int* __restrict__ counts,
                           float* __restrict__ imp, int* __restrict__ cursor) {
    int i = blockIdx.x * blockDim.x + threadIdx.x;
    if (i < out_n) out[i] = 0.0f;
    if (i < MAXPAIRS) { ptok[i] = -1; pprob[i] = 0.0f; }
    if (i < MAXTILES) { meta[2 * i] = -1; meta[2 * i + 1] = 0; }
    if (i < N_EXPERTS) { counts[i] = 0; imp[i] = 0.0f; cursor[i] = 0; }
}

// ---------------- kernel 2: router (one wave32 per token) ----------------
__global__ __launch_bounds__(256) void moe_router_k(
    const float* __restrict__ x, const float* __restrict__ gw,
    const float* __restrict__ gb, int* __restrict__ topk_idx,
    float* __restrict__ topk_prob, int* __restrict__ counts,
    float* __restrict__ imp) {
    const int wave = threadIdx.x >> 5, lane = threadIdx.x & 31;
    const int token = blockIdx.x * 8 + wave;
    const float* xr = x + (size_t)token * D_MODEL;

    float acc[N_EXPERTS];
#pragma unroll
    for (int e = 0; e < N_EXPERTS; ++e) acc[e] = 0.0f;

    for (int d = lane; d < D_MODEL; d += 32) {
        float xv = xr[d];
        const float* g = gw + (size_t)d * N_EXPERTS;
#pragma unroll
        for (int e = 0; e < N_EXPERTS; ++e) acc[e] += xv * g[e];
    }
#pragma unroll
    for (int e = 0; e < N_EXPERTS; ++e)
#pragma unroll
        for (int off = 16; off > 0; off >>= 1)
            acc[e] += __shfl_xor(acc[e], off, 32);

    if (lane == 0) {
        float lg[N_EXPERTS], mx = -1e30f;
#pragma unroll
        for (int e = 0; e < N_EXPERTS; ++e) { lg[e] = acc[e] + gb[e]; mx = fmaxf(mx, lg[e]); }
        float s = 0.0f;
#pragma unroll
        for (int e = 0; e < N_EXPERTS; ++e) s += __expf(lg[e] - mx);
        float inv_s = __builtin_amdgcn_rcpf(s);
#pragma unroll
        for (int e = 0; e < N_EXPERTS; ++e)
            __hip_atomic_fetch_add(&imp[e], __expf(lg[e] - mx) * inv_s,
                                   __ATOMIC_RELAXED, __HIP_MEMORY_SCOPE_AGENT);
        // top-2 (first occurrence wins ties, like jax.lax.top_k)
        int i0 = 0;
#pragma unroll
        for (int e = 1; e < N_EXPERTS; ++e) if (lg[e] > lg[i0]) i0 = e;
        int i1 = (i0 == 0) ? 1 : 0;
#pragma unroll
        for (int e = 0; e < N_EXPERTS; ++e)
            if (e != i0 && lg[e] > lg[i1]) i1 = e;
        float e1 = __expf(lg[i1] - lg[i0]);
        float p0 = __builtin_amdgcn_rcpf(1.0f + e1);
        topk_idx[token * 2]     = i0;
        topk_idx[token * 2 + 1] = i1;
        topk_prob[token * 2]     = p0;
        topk_prob[token * 2 + 1] = e1 * p0;
        atomicAdd(&counts[i0], 1);
        atomicAdd(&counts[i1], 1);
    }
}

// ---------------- kernel 3: per-expert padded offsets + tile metadata ----------------
__global__ void moe_offsets_k(const int* __restrict__ counts,
                              int* __restrict__ cursor, int* __restrict__ meta) {
    if (threadIdx.x == 0 && blockIdx.x == 0) {
        int total = 0, tile = 0;
        for (int e = 0; e < N_EXPERTS; ++e) {
            int cnt = counts[e];
            int padded = (cnt + TILE_M - 1) / TILE_M * TILE_M;
            cursor[e] = total;
            for (int t = 0; t < padded / TILE_M; ++t) {
                meta[2 * tile]     = e;
                meta[2 * tile + 1] = total + t * TILE_M;
                ++tile;
            }
            total += padded;
        }
    }
}

// ---------------- kernel 4: scatter token-expert pairs ----------------
__global__ void moe_scatter_k(const int* __restrict__ topk_idx,
                              const float* __restrict__ topk_prob,
                              int* __restrict__ cursor, int* __restrict__ ptok,
                              float* __restrict__ pprob) {
    int i = blockIdx.x * blockDim.x + threadIdx.x;
    if (i >= N_PAIRS) return;
    int e = topk_idx[i];
    int pos = atomicAdd(&cursor[e], 1);
    ptok[pos]  = i >> 1;          // token id
    pprob[pos] = topk_prob[i];
}

// ---------------- kernel 5: routed expert SwiGLU via WMMA ----------------
__global__ __launch_bounds__(256) void moe_expert_k(
    const float* __restrict__ x,
    const float* __restrict__ w1, const float* __restrict__ b1,
    const float* __restrict__ w2, const float* __restrict__ b2,
    const float* __restrict__ w3, const float* __restrict__ b3,
    const int* __restrict__ meta, const int* __restrict__ ptok,
    const float* __restrict__ pprob, float* __restrict__ out) {
    extern __shared__ char smem[];
    unsigned short* Xs  = (unsigned short*)(smem + LDS_XS);   // 32 x 1032 bf16 (A, GEMM1)
    unsigned short* W1T = (unsigned short*)(smem + LDS_WT);   // 64 x 520 bf16 (B^T, GEMM1)
    unsigned short* W2T = (unsigned short*)(smem + LDS_W2T);  // 64 x 520 bf16
    unsigned short* W3T = (unsigned short*)(smem + LDS_WT);   // 1024 x 72 bf16 (union w/ W1T/W2T)
    unsigned short* Hs  = (unsigned short*)(smem + LDS_HS);   // 32 x 72 bf16 (A, GEMM2)
    __shared__ int   tok_s[TILE_M];
    __shared__ float prob_s[TILE_M];

    const int e = meta[blockIdx.x * 2];
    if (e < 0) return;                               // padding tile (block-uniform)
    const int base = meta[blockIdx.x * 2 + 1];
    const int tid = threadIdx.x;

    if (tid < TILE_M) {
        int t = ptok[base + tid];
        tok_s[tid]  = t;
        prob_s[tid] = (t >= 0) ? pprob[base + tid] : 0.0f;
    }
    __syncthreads();

    // stage gathered X rows into LDS as bf16: b128 global loads, b64 ds stores
    for (int i = tid * 4; i < TILE_M * D_MODEL; i += 256 * 4) {
        int r = i >> 10, c = i & (D_MODEL - 1);
        int t = tok_s[r];
        f32x4 v = (f32x4){0.f, 0.f, 0.f, 0.f};
        if (t >= 0) v = *(const f32x4*)&x[(size_t)t * D_MODEL + c];
        u16x4 p = {f2bf(v.x), f2bf(v.y), f2bf(v.z), f2bf(v.w)};
        *(u16x4*)&Xs[r * XS_STRIDE + c] = p;        // 8B-aligned (c % 4 == 0)
    }

    const int wave = tid >> 5, lane = tid & 31;
    const int ln = lane & 15, hi = lane >> 4;
    const int mt1 = wave >> 2;          // GEMM1 M-tile of this wave (0..1)
    const int nh  = wave & 3;           // GEMM1 N-tile within HK chunk (0..3)

    const float* W1e = w1 + (size_t)e * D_MODEL * D_HIDDEN;
    const float* W2e = w2 + (size_t)e * D_MODEL * D_HIDDEN;
    const float* W3e = w3 + (size_t)e * D_HIDDEN * D_MODEL;
    const float* b1e = b1 + (size_t)e * D_HIDDEN;
    const float* b2e = b2 + (size_t)e * D_HIDDEN;
    const float* b3e = b3 + (size_t)e * D_MODEL;

    // persistent y accumulators: 2 M-tiles x 8 N-tiles per wave (covers N=1024)
    v8f acc[2][8];
#pragma unroll
    for (int mt = 0; mt < 2; ++mt)
#pragma unroll
        for (int j = 0; j < 8; ++j) acc[mt][j] = (v8f){};

    __syncthreads();   // Xs ready

    for (int hc = 0; hc < D_HIDDEN; hc += HK) {
        // ---- GEMM1: h1/h2 tiles, K=1024 in two staged phases of KS=512 ----
        v8f c1 = (v8f){}, c2 = (v8f){};
        const int colh = hc + nh * 16 + ln;          // hidden column of this lane
        for (int ph = 0; ph < 2; ++ph) {
            const int kb = ph * KS;
            // b128 coalesced fp32 reads; transposed bf16 LDS writes (4x b16)
            for (int i = tid * 4; i < KS * HK; i += 256 * 4) {
                int k = i >> 6, c = i & (HK - 1);    // c % 4 == 0, one k per vector
                size_t gi = (size_t)(kb + k) * D_HIDDEN + hc + c;
                __builtin_prefetch(&W1e[gi + HK], 0, 1);    // next chunk's columns
                f32x4 a1 = *(const f32x4*)&W1e[gi];
                f32x4 a2 = *(const f32x4*)&W2e[gi];
                unsigned short* p1 = &W1T[c * W12_STRIDE + k];
                unsigned short* p2 = &W2T[c * W12_STRIDE + k];
                p1[0]               = f2bf(a1.x);
                p1[W12_STRIDE]      = f2bf(a1.y);
                p1[2 * W12_STRIDE]  = f2bf(a1.z);
                p1[3 * W12_STRIDE]  = f2bf(a1.w);
                p2[0]               = f2bf(a2.x);
                p2[W12_STRIDE]      = f2bf(a2.y);
                p2[2 * W12_STRIDE]  = f2bf(a2.z);
                p2[3 * W12_STRIDE]  = f2bf(a2.w);
            }
            __syncthreads();
#pragma unroll 4
            for (int kk = 0; kk < KS; kk += 32) {
                v16bf a   = frag_lds(Xs,  XS_STRIDE,  mt1 * 16 + ln, kb + kk, hi);
                v16bf bw1 = frag_lds(W1T, W12_STRIDE, nh * 16 + ln,  kk,      hi);
                v16bf bw2 = frag_lds(W2T, W12_STRIDE, nh * 16 + ln,  kk,      hi);
                c1 = wmma_bf16(a, bw1, c1);
                c2 = wmma_bf16(a, bw2, c2);
            }
            __syncthreads();   // done reading W1T/W2T before restage
        }
        // fused SwiGLU (v_rcp_f32 instead of IEEE divide), stage h chunk to LDS
        float bb1 = b1e[colh], bb2 = b2e[colh];
#pragma unroll
        for (int r = 0; r < 8; ++r) {
            int m = mt1 * 16 + r + 8 * hi;
            float a1 = c1[r] + bb1;
            float a2 = c2[r] + bb2;
            float h  = a1 * __builtin_amdgcn_rcpf(1.0f + __expf(-a1)) * a2;
            Hs[m * HS_STRIDE + nh * 16 + ln] = f2bf(h);
        }
        // ---- stage w3[hc:hc+64, :] transposed: b128 reads, 4x b16 writes ----
        for (int i = tid * 4; i < HK * D_MODEL; i += 256 * 4) {
            int k = i >> 10, c = i & (D_MODEL - 1);  // c % 4 == 0
            size_t gi = (size_t)(hc + k) * D_MODEL + c;
            __builtin_prefetch(&W3e[gi + (size_t)HK * D_MODEL], 0, 1);
            f32x4 a = *(const f32x4*)&W3e[gi];
            unsigned short* p = &W3T[c * W3_STRIDE + k];
            p[0]              = f2bf(a.x);
            p[W3_STRIDE]      = f2bf(a.y);
            p[2 * W3_STRIDE]  = f2bf(a.z);
            p[3 * W3_STRIDE]  = f2bf(a.w);
        }
        __syncthreads();   // Hs + W3T ready

        // ---- GEMM2: acc += h_chunk(32x64) @ w3[hc:hc+64, :1024] ----
#pragma unroll
        for (int kk2 = 0; kk2 < HK; kk2 += 32) {
            v16bf a0 = frag_lds(Hs, HS_STRIDE, ln,      kk2, hi);
            v16bf a1 = frag_lds(Hs, HS_STRIDE, 16 + ln, kk2, hi);
#pragma unroll
            for (int j = 0; j < 8; ++j) {
                v16bf bw3 = frag_lds(W3T, W3_STRIDE, (wave * 8 + j) * 16 + ln, kk2, hi);
                acc[0][j] = wmma_bf16(a0, bw3, acc[0][j]);
                acc[1][j] = wmma_bf16(a1, bw3, acc[1][j]);
            }
        }
        __syncthreads();   // done reading W3T/Hs before next chunk restages
    }

    // ---- weighted scatter-accumulate into output ----
#pragma unroll
    for (int mt = 0; mt < 2; ++mt) {
#pragma unroll
        for (int j = 0; j < 8; ++j) {
            int colb = (wave * 8 + j) * 16 + ln;
            float bb3 = b3e[colb];
#pragma unroll
            for (int r = 0; r < 8; ++r) {
                int m = mt * 16 + r + 8 * hi;
                int t = tok_s[m];
                if (t >= 0) {
                    float v = prob_s[m] * (acc[mt][j][r] + bb3);
                    __hip_atomic_fetch_add(&out[(size_t)t * D_MODEL + colb], v,
                                           __ATOMIC_RELAXED, __HIP_MEMORY_SCOPE_AGENT);
                }
            }
        }
    }
}

// ---------------- kernel 6: aux load-balance loss ----------------
__global__ void moe_aux_k(const int* __restrict__ counts,
                          const float* __restrict__ imp,
                          float* __restrict__ out, int out_n) {
    if (threadIdx.x == 0 && blockIdx.x == 0) {
        float s = 0.0f;
        for (int e = 0; e < N_EXPERTS; ++e)
            s += (imp[e] / (float)N_TOKENS) * ((float)counts[e] / (float)N_PAIRS);
        out[out_n - 1] = s * (float)N_EXPERTS;
    }
}

// ---------------- launcher ----------------
extern "C" void kernel_launch(void* const* d_in, const int* in_sizes, int n_in,
                              void* d_out, int out_size, void* d_ws, size_t ws_size,
                              hipStream_t stream) {
    const float* x      = (const float*)d_in[0];
    const float* gate_w = (const float*)d_in[1];
    const float* gate_b = (const float*)d_in[2];
    const float* w1     = (const float*)d_in[3];
    const float* b1     = (const float*)d_in[4];
    const float* w2     = (const float*)d_in[5];
    const float* b2     = (const float*)d_in[6];
    const float* w3     = (const float*)d_in[7];
    const float* b3     = (const float*)d_in[8];
    float* out = (float*)d_out;

    // workspace layout
    char* ws = (char*)d_ws;
    int*   topk_idx  = (int*)  (ws);                                  // 64KB
    float* topk_prob = (float*)(ws + 65536);                          // 64KB
    int*   counts    = (int*)  (ws + 131072);
    float* imp       = (float*)(ws + 131072 + 64);
    int*   cursor    = (int*)  (ws + 131072 + 128);
    int*   ptok      = (int*)  (ws + 131072 + 256);
    float* pprob     = (float*)(ws + 131072 + 256 + MAXPAIRS * 4);
    int*   meta      = (int*)  (ws + 131072 + 256 + MAXPAIRS * 8);

    // 1. init outputs + workspace sentinels
    moe_init_k<<<(out_size + 255) / 256, 256, 0, stream>>>(out, out_size, ptok,
                                                           pprob, meta, counts,
                                                           imp, cursor);
    // 2. router: one wave per token, 8 tokens per block
    moe_router_k<<<N_TOKENS / 8, 256, 0, stream>>>(x, gate_w, gate_b, topk_idx,
                                                   topk_prob, counts, imp);
    // 3. padded per-expert offsets + tile metadata
    moe_offsets_k<<<1, 32, 0, stream>>>(counts, cursor, meta);
    // 4. scatter pairs into per-expert segments
    moe_scatter_k<<<(N_PAIRS + 255) / 256, 256, 0, stream>>>(topk_idx, topk_prob,
                                                             cursor, ptok, pprob);
    // 5. routed expert SwiGLU (grid ordered expert-major -> L2 weight reuse)
    moe_expert_k<<<MAXTILES, 256, LDS_TOTAL, stream>>>(x, w1, b1, w2, b2, w3, b3,
                                                       meta, ptok, pprob, out);
    // 6. aux loss scalar
    moe_aux_k<<<1, 32, 0, stream>>>(counts, imp, out, out_size);
}
